// FastWeightGluMLPMultihead_58712202936371
// MI455X (gfx1250) — matrix operand order repaired and checked by reading, not compiled, not run on hardware
//
#include <hip/hip_runtime.h>
#include <math.h>

// Problem constants (match reference)
#define BB   16
#define LL   4096
#define DD   256
#define DHH  512
#define CHN  1024
#define NCH  4      // LL / CHN

typedef __attribute__((ext_vector_type(16))) __bf16 v16bf;
typedef __attribute__((ext_vector_type(8)))  __bf16 v8bf;
typedef __attribute__((ext_vector_type(4)))  __bf16 v4bf;
typedef __attribute__((ext_vector_type(8)))  float  v8f;
typedef __attribute__((ext_vector_type(4)))  float  v4f;

__device__ inline __bf16 bf(float x) { return (__bf16)x; }
__device__ inline float  ff(__bf16 x) { return (float)x; }

__device__ inline v8f wmma_bf16(v16bf a, v16bf b, v8f c) {
  return __builtin_amdgcn_wmma_f32_16x16x32_bf16(
      false, a, false, b, (short)0, c, false, false);
}

// ---------------------------------------------------------------------
// Async global->LDS 16-byte copy (CDNA5 GLOBAL_LOAD_ASYNC_TO_LDS_B128,
// tracked by ASYNCcnt). LDS byte address = low 32 bits of the generic
// shared pointer (flat LDS aperture mapping, ISA 10.2).
// ---------------------------------------------------------------------
__device__ inline void async_copy_b128(__bf16* lds_dst, const __bf16* gsrc) {
  const unsigned l = (unsigned)(unsigned long long)lds_dst;
  asm volatile("global_load_async_to_lds_b128 %0, %1, off"
               :: "v"(l), "v"(gsrc)
               : "memory");
}
__device__ inline void async_wait0() {
  asm volatile("s_wait_asynccnt 0x0" ::: "memory");
}

// ---------------------------------------------------------------------
// Wide fragment builders (wave32, CDNA5 ISA 7.12.2 layouts).
// ---------------------------------------------------------------------
__device__ inline v16bf frag_a_w(const __bf16* p, int ld, int m0, int k0) {
  const int lane = threadIdx.x & 31;
  const int row = m0 + (lane & 15);
  const int kb  = k0 + ((lane >> 4) << 3);
  const __bf16* base = p + row * ld + kb;
  const v8bf lo = *(const v8bf*)(base);
  const v8bf hi = *(const v8bf*)(base + 16);
  v16bf f;
#pragma unroll
  for (int i = 0; i < 8; ++i) { f[i] = lo[i]; f[i + 8] = hi[i]; }
  return f;
}

// B frag (32x16) from K-contiguous layout: B[k,n] = p[n*ld + k].
__device__ inline v16bf frag_bt_w(const __bf16* p, int ld, int k0, int n0) {
  const int lane = threadIdx.x & 31;
  const int n  = n0 + (lane & 15);
  const int kb = k0 + ((lane >> 4) << 3);
  const __bf16* base = p + n * ld + kb;
  const v8bf lo = *(const v8bf*)(base);
  const v8bf hi = *(const v8bf*)(base + 16);
  v16bf f;
#pragma unroll
  for (int i = 0; i < 8; ++i) { f[i] = lo[i]; f[i + 8] = hi[i]; }
  return f;
}

__device__ inline void store_c_bf16(__bf16* p, int ld, int m0, int n0, v8f c) {
  const int lane = threadIdx.x & 31;
  const int col = n0 + (lane & 15);
  const int mb  = m0 + ((lane >> 4) << 3);
#pragma unroll
  for (int r = 0; r < 8; ++r) p[(mb + r) * ld + col] = bf(c[r]);
}

__device__ inline void store_c_bf16_T(__bf16* p, int ldm, int m0, int n0, v8f c) {
  const int lane = threadIdx.x & 31;
  const int col = n0 + (lane & 15);
  const int mb  = m0 + ((lane >> 4) << 3);
  v8bf v;
#pragma unroll
  for (int r = 0; r < 8; ++r) v[r] = bf(c[r]);
  *(v8bf*)(p + (long)col * ldm + mb) = v;
}

__device__ inline void store_c_f32_T(float* p, int ldm, int m0, int n0, v8f c) {
  const int lane = threadIdx.x & 31;
  const int col = n0 + (lane & 15);
  const int mb  = m0 + ((lane >> 4) << 3);
  v4f lo = {c[0], c[1], c[2], c[3]};
  v4f hi = {c[4], c[5], c[6], c[7]};
  *(v4f*)(p + (long)col * ldm + mb)     = lo;
  *(v4f*)(p + (long)col * ldm + mb + 4) = hi;
}

__device__ inline v8f load_c_bf16_T(const __bf16* p, int ldm, int m0, int n0) {
  const int lane = threadIdx.x & 31;
  const int col = n0 + (lane & 15);
  const int mb  = m0 + ((lane >> 4) << 3);
  const v8bf v = *(const v8bf*)(p + (long)col * ldm + mb);
  v8f o;
#pragma unroll
  for (int r = 0; r < 8; ++r) o[r] = ff(v[r]);
  return o;
}

// =====================================================================
// K_INIT: weights -> transposed masters + norms + bf16 mirrors. grid (B,3)
// =====================================================================
__global__ __launch_bounds__(256) void k_init(
    const float* __restrict__ w0i, const float* __restrict__ w1i,
    const float* __restrict__ w2i,
    float* __restrict__ wW0T, float* __restrict__ wW1T, float* __restrict__ wW2T,
    float* __restrict__ n0a, float* __restrict__ n1a, float* __restrict__ n2a,
    __bf16* __restrict__ w0T, __bf16* __restrict__ w2T,
    __bf16* __restrict__ w1b, __bf16* __restrict__ w1bT)
{
  const int b = blockIdx.x, w = blockIdx.y;
  const long wo = (long)b * (DD * DHH);
  if (w != 1) {
    const float* S = ((w == 0) ? w0i : w2i) + wo;    // [256,512]
    float* Wd      = ((w == 0) ? wW0T : wW2T) + wo;  // [512,256]
    float* nn      = ((w == 0) ? n0a : n2a) + (long)b * DHH;
    __bf16* Mb     = ((w == 0) ? w0T : w2T) + wo;    // [512,256]
    for (int h = threadIdx.x; h < DHH; h += 256) {
      float ss = 0.f;
      for (int d = 0; d < DD; ++d) {
        const float v = S[d * DHH + h];
        Wd[h * DD + d] = v;
        Mb[h * DD + d] = bf(v);
        ss += v * v;
      }
      nn[h] = sqrtf(ss);
    }
  } else {
    const float* S = w1i + wo;       // [512,256]
    float* Wd = wW1T + wo;           // [256,512]
    float* nn = n1a + (long)b * DD;
    __bf16* Mb  = w1b  + wo;         // [512,256]
    __bf16* MbT = w1bT + wo;         // [256,512]
    for (int d = threadIdx.x; d < DD; d += 256) {
      float ss = 0.f;
      for (int h = 0; h < DHH; ++h) {
        const float v = S[h * DD + d];
        Wd[d * DHH + h]  = v;
        Mb[h * DD + d]   = bf(v);
        MbT[d * DHH + h] = bf(v);
        ss += v * v;
      }
      nn[d] = sqrtf(ss);
    }
  }
}

// =====================================================================
// K_S1: gate=k@w0, hbm=k@w2, dh=(-v)@w1^T + SiLU fwd/bwd epilogue.
// Weight tiles staged with async b128 loads. grid (B,16,4), 256 thr.
// =====================================================================
__global__ __launch_bounds__(256) void k_s1(
    const float* __restrict__ kg, const float* __restrict__ vg,
    const float* __restrict__ lr1g,
    const __bf16* __restrict__ w0T, const __bf16* __restrict__ w2T,
    const __bf16* __restrict__ w1b,
    __bf16* __restrict__ hl1T, __bf16* __restrict__ dgbaT,
    __bf16* __restrict__ dhbmT, int chunk)
{
  __shared__ __align__(16) __bf16 sK[64 * 32];    // [c][d]
  __shared__ __align__(16) __bf16 sV[64 * 32];
  __shared__ __align__(16) __bf16 sW0[128 * 32];  // [h][d]  (K-contig)
  __shared__ __align__(16) __bf16 sW2[128 * 32];
  __shared__ __align__(16) __bf16 sW1[128 * 32];
  __shared__ float l1row[64];
  const int b = blockIdx.x, m0 = blockIdx.y * 64, n0 = blockIdx.z * 128;
  const int tid = threadIdx.x, lane = tid & 31, wave = tid >> 5;
  const int wm = wave & 1, wn = wave >> 1;
  const long rowbase = (long)b * LL + (long)chunk * CHN + m0;
  if (tid < 64) l1row[tid] = lr1g[rowbase + tid];
  v8f aG[2][2] = {}, aH[2][2] = {}, aD[2][2] = {};

  for (int d0 = 0; d0 < DD; d0 += 32) {
    __syncthreads();
    for (int i = tid; i < 512; i += 256) {          // A tiles: 64x32 / 4
      const int r = i >> 3, c4 = (i & 7) << 2;
      const long src = (rowbase + r) * DD + d0 + c4;
      const v4f kv = *(const v4f*)(kg + src);
      const v4f vv = *(const v4f*)(vg + src);
      v4bf k4, v4;
#pragma unroll
      for (int t = 0; t < 4; ++t) { k4[t] = bf(kv[t]); v4[t] = bf(-vv[t]); }
      *(v4bf*)(sK + r * 32 + c4) = k4;
      *(v4bf*)(sV + r * 32 + c4) = v4;
    }
    for (int i = tid; i < 512; i += 256) {          // W tiles: 128x32 / 8 async
      const int r = i >> 2, c8 = (i & 3) << 3;
      const long src = ((long)b * DHH + n0 + r) * DD + d0 + c8;
      const int dst = r * 32 + c8;
      async_copy_b128(sW0 + dst, w0T + src);
      async_copy_b128(sW2 + dst, w2T + src);
      async_copy_b128(sW1 + dst, w1b + src);
    }
    async_wait0();
    __syncthreads();
    const v16bf aK0 = frag_a_w(sK, 32, wm * 32,      0);
    const v16bf aK1 = frag_a_w(sK, 32, wm * 32 + 16, 0);
    const v16bf aV0 = frag_a_w(sV, 32, wm * 32,      0);
    const v16bf aV1 = frag_a_w(sV, 32, wm * 32 + 16, 0);
#pragma unroll
    for (int j = 0; j < 2; ++j) {
      const v16bf b0 = frag_bt_w(sW0, 32, 0, wn * 32 + j * 16);
      const v16bf b2 = frag_bt_w(sW2, 32, 0, wn * 32 + j * 16);
      const v16bf b1 = frag_bt_w(sW1, 32, 0, wn * 32 + j * 16);
      aG[0][j] = wmma_bf16(aK0, b0, aG[0][j]);
      aG[1][j] = wmma_bf16(aK1, b0, aG[1][j]);
      aH[0][j] = wmma_bf16(aK0, b2, aH[0][j]);
      aH[1][j] = wmma_bf16(aK1, b2, aH[1][j]);
      aD[0][j] = wmma_bf16(aV0, b1, aD[0][j]);
      aD[1][j] = wmma_bf16(aV1, b1, aD[1][j]);
    }
  }
#pragma unroll
  for (int i = 0; i < 2; ++i) {
#pragma unroll
    for (int j = 0; j < 2; ++j) {
      const int mb  = wm * 32 + i * 16 + ((lane >> 4) << 3);
      const int nn2 = n0 + wn * 32 + j * 16 + (lane & 15);
      v8bf h8, g8, d8;
#pragma unroll
      for (int r = 0; r < 8; ++r) {
        const int ml = mb + r;
        const float g   = aG[i][j][r];
        const float hb  = aH[i][j][r];
        const float dh  = aD[i][j][r];
        const float s   = 1.f / (1.f + __expf(-g));
        const float sil = g * s;
        const float dgate = dh * hb;
        h8[r] = bf(sil * hb * l1row[ml]);
        g8[r] = bf(dgate * s * (1.f + g * (1.f - s)));
        d8[r] = bf(dh * sil);
      }
      const long o = ((long)b * DHH + nn2) * CHN + m0 + mb; // transposed [h][c]
      *(v8bf*)(hl1T  + o) = h8;
      *(v8bf*)(dgbaT + o) = g8;
      *(v8bf*)(dhbmT + o) = d8;
    }
  }
}

// =====================================================================
// K_S2: gradient GEMMs C[d,h] = sum_c P[c,d]*Q[c,h]  (K=1024).
// 64x64 wave tiles (16 accum tiles) -> 8 frag loads per 16 WMMAs.
// Q staged with async b128. grid (B,3,4), 256 thr, block tile 128x256.
// =====================================================================
__global__ __launch_bounds__(256) void k_s2(
    const float* __restrict__ kg, const float* __restrict__ vg,
    const float* __restrict__ lr0g, const float* __restrict__ lr2g,
    const __bf16* __restrict__ dgbaT, const __bf16* __restrict__ hl1T,
    const __bf16* __restrict__ dhbmT,
    float* __restrict__ grads, int chunk)
{
  __shared__ __align__(16) __bf16 sPt[128 * 32];   // [d][c]
  __shared__ __align__(16) __bf16 sQt[256 * 32];   // [h][c]
  const int b = blockIdx.x, w = blockIdx.y;
  const int m0 = (blockIdx.z >> 1) * 128, n0 = (blockIdx.z & 1) * 256;
  const int tid = threadIdx.x, wave = tid >> 5;
  const int wm = wave & 1, wn = wave >> 1;
  const __bf16* Qt = (w == 0) ? dgbaT : (w == 1) ? hl1T : dhbmT;
  const float* lrp = (w == 0) ? lr0g : lr2g;
  v8f acc[4][4] = {};

  for (int c0 = 0; c0 < CHN; c0 += 32) {
    __syncthreads();
    for (int i = tid; i < 1024; i += 256) {       // 32(c) x 128(d) / 4
      const int r = i >> 5, d4 = (i & 31) << 2;
      const long row = (long)b * LL + (long)chunk * CHN + c0 + r;
      if (w == 1) {
        const v4f pv = *(const v4f*)(vg + row * DD + m0 + d4);
#pragma unroll
        for (int t = 0; t < 4; ++t) sPt[(d4 + t) * 32 + r] = bf(-pv[t]);
      } else {
        const v4f pv = *(const v4f*)(kg + row * DD + m0 + d4);
        const float sc = lrp[row];
#pragma unroll
        for (int t = 0; t < 4; ++t) sPt[(d4 + t) * 32 + r] = bf(pv[t] * sc);
      }
    }
    for (int i = tid; i < 1024; i += 256) {       // 256(h) x 32(c) / 8 async
      const int r = i >> 2, c8 = (i & 3) << 3;
      async_copy_b128(sQt + r * 32 + c8,
                      Qt + ((long)b * DHH + n0 + r) * CHN + c0 + c8);
    }
    async_wait0();
    __syncthreads();
    v16bf af[4];
#pragma unroll
    for (int i = 0; i < 4; ++i)
      af[i] = frag_a_w(sPt, 32, wm * 64 + i * 16, 0);
#pragma unroll
    for (int j = 0; j < 4; ++j) {
      const v16bf bq = frag_bt_w(sQt, 32, 0, wn * 64 + j * 16);
#pragma unroll
      for (int i = 0; i < 4; ++i) acc[i][j] = wmma_bf16(af[i], bq, acc[i][j]);
    }
  }
  float* Gt = grads + (long)(b * 3 + w) * (DD * DHH);   // [512,256]
#pragma unroll
  for (int i = 0; i < 4; ++i)
#pragma unroll
    for (int j = 0; j < 4; ++j)
      store_c_f32_T(Gt, DD, m0 + wm * 64 + i * 16, n0 + wn * 64 + j * 16,
                    acc[i][j]);
}

// =====================================================================
// K_NS: quintic Newton-Schulz, 5 steps bf16. Macro-tiled phases
// (64x64 per wave). X [256,512] + Xt; A symmetric; Yt. grid (48), 512 thr.
// =====================================================================
__global__ __launch_bounds__(512) void k_ns(
    const float* __restrict__ grads,
    __bf16* __restrict__ Xg, __bf16* __restrict__ XtG,
    __bf16* __restrict__ Ag, __bf16* __restrict__ YtG)
{
  __shared__ float red[17];
  const int p = blockIdx.x;
  const int tid = threadIdx.x, lane = tid & 31, wave = tid >> 5;
  const int wm4 = wave >> 2, wn4 = wave & 3;        // 4x4 macro grid
  const float* Gt = grads + (long)p * (DD * DHH);   // [512,256]
  __bf16* X  = Xg  + (long)p * (DD * DHH);          // [256,512]
  __bf16* Xt = XtG + (long)p * (DD * DHH);          // [512,256]
  __bf16* A  = Ag  + (long)p * (DD * DD);           // [256,256] symmetric
  __bf16* Yt = YtG + (long)p * (DD * DHH);          // [512,256]

  float ss = 0.f;
  for (int j = tid; j < DD * DHH; j += 512) {       // j = h*256 + d
    const __bf16 xv = bf(Gt[j]);
    Xt[j] = xv;
    X[(j & 255) * DHH + (j >> 8)] = xv;
    const float f = ff(xv);
    ss += f * f;
  }
#pragma unroll
  for (int o = 16; o > 0; o >>= 1) ss += __shfl_down(ss, o, 32);
  if (lane == 0) red[wave] = ss;
  __syncthreads();
  if (tid == 0) {
    float t = 0.f;
    for (int i = 0; i < 16; ++i) t += red[i];
    red[16] = 1.f / (sqrtf(t) + 1e-7f);
  }
  __syncthreads();
  const float sc = red[16];
  for (int j = tid; j < DD * DHH; j += 512) Xt[j] = bf(ff(Xt[j]) * sc);
  for (int i = tid; i < DD * DHH; i += 512) X[i]  = bf(ff(X[i])  * sc);
  __threadfence();
  __syncthreads();

  const float ca = 3.4445f, cb = -4.775f, cc = 2.0315f;
  for (int step = 0; step < 5; ++step) {
    // phase 1: A = X X^T (256x256, K=512); one 64x64 macro per wave
    {
      const int tm = wm4 * 64, tn = wn4 * 64;
      v8f acc[4][4] = {};
      for (int k0 = 0; k0 < DHH; k0 += 32) {
        v16bf af[4];
#pragma unroll
        for (int i = 0; i < 4; ++i) af[i] = frag_a_w(X, DHH, tm + i * 16, k0);
#pragma unroll
        for (int j = 0; j < 4; ++j) {
          const v16bf bfr = frag_bt_w(X, DHH, k0, tn + j * 16);
#pragma unroll
          for (int i = 0; i < 4; ++i)
            acc[i][j] = wmma_bf16(af[i], bfr, acc[i][j]);
        }
      }
#pragma unroll
      for (int i = 0; i < 4; ++i)
#pragma unroll
        for (int j = 0; j < 4; ++j)
          store_c_bf16_T(A, DD, tm + i * 16, tn + j * 16, acc[i][j]);
    }
    __threadfence();
    __syncthreads();
    // phase 2: Y = A X (256x512, K=256); two 64x64 macros per wave
    for (int half = 0; half < 2; ++half) {
      const int tm = wm4 * 64, tn = (wn4 + half * 4) * 64;
      v8f acc[4][4] = {};
      for (int k0 = 0; k0 < DD; k0 += 32) {
        v16bf af[4];
#pragma unroll
        for (int i = 0; i < 4; ++i) af[i] = frag_a_w(A, DD, tm + i * 16, k0);
#pragma unroll
        for (int j = 0; j < 4; ++j) {
          const v16bf bfr = frag_bt_w(Xt, DD, k0, tn + j * 16);
#pragma unroll
          for (int i = 0; i < 4; ++i)
            acc[i][j] = wmma_bf16(af[i], bfr, acc[i][j]);
        }
      }
#pragma unroll
      for (int i = 0; i < 4; ++i)
#pragma unroll
        for (int j = 0; j < 4; ++j)
          store_c_bf16_T(Yt, DD, tm + i * 16, tn + j * 16, acc[i][j]);
    }
    __threadfence();
    __syncthreads();
    // phase 3: X = a*X + b*Y + c*(A Y); own-tile X update only
    for (int half = 0; half < 2; ++half) {
      const int tm = wm4 * 64, tn = (wn4 + half * 4) * 64;
      v8f acc[4][4] = {};
      for (int k0 = 0; k0 < DD; k0 += 32) {
        v16bf af[4];
#pragma unroll
        for (int i = 0; i < 4; ++i) af[i] = frag_a_w(A, DD, tm + i * 16, k0);
#pragma unroll
        for (int j = 0; j < 4; ++j) {
          const v16bf bfr = frag_bt_w(Yt, DD, k0, tn + j * 16);
#pragma unroll
          for (int i = 0; i < 4; ++i)
            acc[i][j] = wmma_bf16(af[i], bfr, acc[i][j]);
        }
      }
#pragma unroll
      for (int i = 0; i < 4; ++i)
#pragma unroll
        for (int j = 0; j < 4; ++j) {
          const int tmi = tm + i * 16, tnj = tn + j * 16;
          const v8f xv = load_c_bf16_T(Xt, DD, tmi, tnj);
          const v8f yv = load_c_bf16_T(Yt, DD, tmi, tnj);
          v8f nx;
#pragma unroll
          for (int r = 0; r < 8; ++r)
            nx[r] = ca * xv[r] + cb * yv[r] + cc * acc[i][j][r];
          store_c_bf16  (X,  DHH, tmi, tnj, nx);
          store_c_bf16_T(Xt, DD,  tmi, tnj, nx);
        }
    }
    __threadfence();
    __syncthreads();
  }
}

// =====================================================================
// K_UPD: w -= NS(grad); per-column renorm; refresh bf16 mirrors. grid (B,3)
// =====================================================================
__global__ __launch_bounds__(256) void k_upd(
    const __bf16* __restrict__ Xg, const __bf16* __restrict__ XtG,
    float* __restrict__ wW0T, float* __restrict__ wW1T, float* __restrict__ wW2T,
    const float* __restrict__ n0a, const float* __restrict__ n1a,
    const float* __restrict__ n2a,
    __bf16* __restrict__ w0T, __bf16* __restrict__ w2T,
    __bf16* __restrict__ w1b, __bf16* __restrict__ w1bT)
{
  const int b = blockIdx.x, w = blockIdx.y;
  const long po = (long)(b * 3 + w) * (DD * DHH);
  const long wo = (long)b * (DD * DHH);
  if (w != 1) {
    const __bf16* Xt = XtG + po;                     // [512,256]
    float* Wd  = ((w == 0) ? wW0T : wW2T) + wo;      // [512,256]
    const float* nn = ((w == 0) ? n0a : n2a) + (long)b * DHH;
    __bf16* Mb = ((w == 0) ? w0T : w2T) + wo;
    for (int h = threadIdx.x; h < DHH; h += 256) {
      float ssq = 0.f;
      for (int d = 0; d < DD; ++d) {
        const float v = Wd[h * DD + d] - ff(Xt[h * DD + d]);
        ssq += v * v;
        Wd[h * DD + d] = v;
      }
      const float s = nn[h] / (sqrtf(ssq) + 1e-5f);
      for (int d = 0; d < DD; ++d) {
        const float v = Wd[h * DD + d] * s;
        Wd[h * DD + d] = v;
        Mb[h * DD + d] = bf(v);
      }
    }
  } else {
    const __bf16* X = Xg + po;                       // [256,512] = NS(w1g)^T
    float* Wd = wW1T + wo;                           // [256,512] (w1^T master)
    const float* nn = n1a + (long)b * DD;
    __bf16* Mb  = w1b  + wo;                         // [512,256]
    __bf16* MbT = w1bT + wo;                         // [256,512]
    for (int d = threadIdx.x; d < DD; d += 256) {
      float ssq = 0.f;
      for (int h = 0; h < DHH; ++h) {
        const float v = Wd[d * DHH + h] - ff(X[d * DHH + h]);
        ssq += v * v;
        Wd[d * DHH + h] = v;
      }
      const float s = nn[d] / (sqrtf(ssq) + 1e-5f);
      for (int h = 0; h < DHH; ++h) {
        const float v = Wd[d * DHH + h] * s;
        Wd[d * DHH + h]  = v;
        MbT[d * DHH + h] = bf(v);
        Mb[h * DD + d]   = bf(v);
      }
    }
  }
}

// =====================================================================
// K_S3A: hq = silu(q@w0)*(q@w2) -> hqT [B,DH,CH] bf16. grid (B,16,4)
// =====================================================================
__global__ __launch_bounds__(256) void k_s3a(
    const float* __restrict__ qg,
    const __bf16* __restrict__ w0T, const __bf16* __restrict__ w2T,
    __bf16* __restrict__ hqT, int chunk)
{
  __shared__ __align__(16) __bf16 sQ[64 * 32];
  __shared__ __align__(16) __bf16 sW0[128 * 32];
  __shared__ __align__(16) __bf16 sW2[128 * 32];
  const int b = blockIdx.x, m0 = blockIdx.y * 64, n0 = blockIdx.z * 128;
  const int tid = threadIdx.x, lane = tid & 31, wave = tid >> 5;
  const int wm = wave & 1, wn = wave >> 1;
  const long rowbase = (long)b * LL + (long)chunk * CHN + m0;
  v8f aG[2][2] = {}, aH[2][2] = {};

  for (int d0 = 0; d0 < DD; d0 += 32) {
    __syncthreads();
    for (int i = tid; i < 512; i += 256) {
      const int r = i >> 3, c4 = (i & 7) << 2;
      const v4f qv = *(const v4f*)(qg + (rowbase + r) * DD + d0 + c4);
      v4bf q4;
#pragma unroll
      for (int t = 0; t < 4; ++t) q4[t] = bf(qv[t]);
      *(v4bf*)(sQ + r * 32 + c4) = q4;
    }
    for (int i = tid; i < 512; i += 256) {
      const int r = i >> 2, c8 = (i & 3) << 3;
      const long src = ((long)b * DHH + n0 + r) * DD + d0 + c8;
      const int dst = r * 32 + c8;
      async_copy_b128(sW0 + dst, w0T + src);
      async_copy_b128(sW2 + dst, w2T + src);
    }
    async_wait0();
    __syncthreads();
    const v16bf a0 = frag_a_w(sQ, 32, wm * 32,      0);
    const v16bf a1 = frag_a_w(sQ, 32, wm * 32 + 16, 0);
#pragma unroll
    for (int j = 0; j < 2; ++j) {
      const v16bf b0 = frag_bt_w(sW0, 32, 0, wn * 32 + j * 16);
      const v16bf b2 = frag_bt_w(sW2, 32, 0, wn * 32 + j * 16);
      aG[0][j] = wmma_bf16(a0, b0, aG[0][j]);
      aG[1][j] = wmma_bf16(a1, b0, aG[1][j]);
      aH[0][j] = wmma_bf16(a0, b2, aH[0][j]);
      aH[1][j] = wmma_bf16(a1, b2, aH[1][j]);
    }
  }
#pragma unroll
  for (int i = 0; i < 2; ++i)
#pragma unroll
    for (int j = 0; j < 2; ++j) {
      const int mb  = wm * 32 + i * 16 + ((lane >> 4) << 3);
      const int nn2 = n0 + wn * 32 + j * 16 + (lane & 15);
      v8bf h8;
#pragma unroll
      for (int r = 0; r < 8; ++r) {
        const float g  = aG[i][j][r];
        const float hb = aH[i][j][r];
        const float s  = 1.f / (1.f + __expf(-g));
        h8[r] = bf(g * s * hb);
      }
      *(v8bf*)(hqT + ((long)b * DHH + nn2) * CHN + m0 + mb) = h8;
    }
}

// =====================================================================
// K_S3B: O[c,d] = sum_h hq[c,h]*w1[h,d] computed as C[m=d,n=c] so the
// f32 store into row-major o is contiguous. grid (B,8,4), 256 thr
// =====================================================================
__global__ __launch_bounds__(256) void k_s3b(
    const __bf16* __restrict__ hqT, const __bf16* __restrict__ w1bT,
    float* __restrict__ out, int chunk)
{
  __shared__ __align__(16) __bf16 sA[64 * 32];   // [d][h]
  __shared__ __align__(16) __bf16 sB[128 * 32];  // [c][h]
  const int b = blockIdx.x, n0 = blockIdx.y * 128, m0 = blockIdx.z * 64;
  const int tid = threadIdx.x, wave = tid >> 5;
  const int wm = wave & 1, wn = wave >> 1;
  v8f acc[2][2] = {};

  for (int k0 = 0; k0 < DHH; k0 += 32) {
    __syncthreads();
    {                                             // 64(d) x 32(h) / 8 async
      const int i = tid;
      if (i < 256) {
        const int r = i >> 2, c8 = (i & 3) << 3;
        const long src = ((long)b * DD + m0 + r) * DHH + k0 + c8;
        async_copy_b128(sA + r * 32 + c8, w1bT + src);
      }
    }
    for (int i = tid; i < 512; i += 256) {        // 32(h) x 128(c) / 8
      const int hh = i >> 4, c8 = (i & 15) << 3;
      const long src = ((long)b * DHH + k0 + hh) * CHN + n0 + c8;
      const v8bf qv = *(const v8bf*)(hqT + src);
#pragma unroll
      for (int t = 0; t < 8; ++t) sB[(c8 + t) * 32 + hh] = qv[t];
    }
    async_wait0();
    __syncthreads();
    const v16bf a0 = frag_a_w(sA, 32, wm * 32,      0);
    const v16bf a1 = frag_a_w(sA, 32, wm * 32 + 16, 0);
#pragma unroll
    for (int j = 0; j < 2; ++j) {
      const v16bf bq = frag_bt_w(sB, 32, 0, wn * 32 + j * 16);
      acc[0][j] = wmma_bf16(a0, bq, acc[0][j]);
      acc[1][j] = wmma_bf16(a1, bq, acc[1][j]);
    }
  }
  float* Og = out + (long)b * LL * DD + (long)chunk * CHN * DD;  // [c][d]
#pragma unroll
  for (int i = 0; i < 2; ++i)
#pragma unroll
    for (int j = 0; j < 2; ++j)
      store_c_f32_T(Og, DD, m0 + wm * 32 + i * 16, n0 + wn * 32 + j * 16,
                    acc[i][j]);
}

// =====================================================================
// Host launcher
// =====================================================================
extern "C" void kernel_launch(void* const* d_in, const int* in_sizes, int n_in,
                              void* d_out, int out_size, void* d_ws, size_t ws_size,
                              hipStream_t stream) {
  (void)in_sizes; (void)n_in; (void)out_size; (void)ws_size;
  const float* q   = (const float*)d_in[0];
  const float* k   = (const float*)d_in[1];
  const float* v   = (const float*)d_in[2];
  const float* lr0 = (const float*)d_in[3];
  const float* lr1 = (const float*)d_in[4];
  const float* lr2 = (const float*)d_in[5];
  const float* w0i = (const float*)d_in[6];
  const float* w1i = (const float*)d_in[7];
  const float* w2i = (const float*)d_in[8];
  float* out = (float*)d_out;

  const long SZW = (long)BB * DD * DHH;            // 2,097,152 elems
  float* wW0T = (float*)d_ws;
  float* wW1T = wW0T + SZW;
  float* wW2T = wW1T + SZW;
  float* n0a  = wW2T + SZW;
  float* n1a  = n0a + (long)BB * DHH;
  float* n2a  = n1a + (long)BB * DD;
  float* grads = n2a + (long)BB * DHH;             // 48 x [512,256] f32
  __bf16* w0T  = (__bf16*)(grads + 48L * DD * DHH);
  __bf16* w2T  = w0T  + SZW;
  __bf16* w1b  = w2T  + SZW;                       // [B,512,256]
  __bf16* w1bT = w1b  + SZW;                       // [B,256,512]
  __bf16* hl1T  = w1bT + SZW;                      // [B,DH,CH]
  __bf16* dgbaT = hl1T  + (long)BB * CHN * DHH;
  __bf16* dhbmT = dgbaT + (long)BB * CHN * DHH;
  __bf16* Xg  = dhbmT + (long)BB * CHN * DHH;      // 48 x [256,512]
  __bf16* XtG = Xg  + 48L * DD * DHH;              // 48 x [512,256]
  __bf16* Ag  = XtG + 48L * DD * DHH;              // 48 x [256,256]
  __bf16* YtG = Ag  + 48L * DD * DD;               // 48 x [512,256]
  __bf16* hqT = hl1T;  // alias: hl1T dead after k_s2, reused for stage 3

  k_init<<<dim3(BB, 3), 256, 0, stream>>>(w0i, w1i, w2i, wW0T, wW1T, wW2T,
                                          n0a, n1a, n2a, w0T, w2T, w1b, w1bT);
  for (int t = 0; t < NCH; ++t) {
    k_s1 <<<dim3(BB, 16, 4), 256, 0, stream>>>(k, v, lr1, w0T, w2T, w1b,
                                               hl1T, dgbaT, dhbmT, t);
    k_s2 <<<dim3(BB, 3, 4), 256, 0, stream>>>(k, v, lr0, lr2,
                                              dgbaT, hl1T, dhbmT, grads, t);
    k_ns <<<dim3(48), 512, 0, stream>>>(grads, Xg, XtG, Ag, YtG);
    k_upd<<<dim3(BB, 3), 256, 0, stream>>>(Xg, XtG, wW0T, wW1T, wW2T,
                                           n0a, n1a, n2a, w0T, w2T, w1b, w1bT);
    k_s3a<<<dim3(BB, 16, 4), 256, 0, stream>>>(q, w0T, w2T, hqT, t);
    k_s3b<<<dim3(BB, 8, 4), 256, 0, stream>>>(hqT, w1bT, out, t);
  }
}
